// SVGMDNTop_77884936945867
// MI455X (gfx1250) — matrix operand order repaired and checked by reading, not compiled, not run on hardware
//
#include <hip/hip_runtime.h>
#include <hip/hip_bf16.h>

// ---------------------------------------------------------------------------
// Fused  [dec(104448x1024) x W^T(1024x904) + b]  -> cmd-argmax one-hot (4)
//        + per-group mixture argmax gather + gaussian (6)  => out (104448x10)
// Pass 0 (tiny): W f32 -> bf16 into d_ws (1.85MB, L2-resident).
// Pass 1 (fused): bf16 WMMA GEMM + sampling epilogue; A converted on the fly
//                 (exactly once per element), A-stage double-buffered so the
//                 next chunk's HBM latency overlaps the current chunk's WMMAs.
// ---------------------------------------------------------------------------

typedef __attribute__((ext_vector_type(16))) __bf16 v16bf;
typedef __attribute__((ext_vector_type(8)))  float  v8f;

#define SDIM    51
#define BDIM    2048
#define HDIM    1024
#define OUTC    904        // 4 + 6*50*3
#define NMIX    50
#define BLOCK_M 32
#define CHUNK_K 128
#define NCHUNK  (HDIM / CHUNK_K)
#define APITCH  136        // bf16 elems per staged A row (pad vs bank conflicts)
#define ABUFSZ  (BLOCK_M * APITCH)
#define CPITCH  912        // f32 elems per C row in LDS (57 tiles * 16)
#define NTILES  57
#define NWAVES  16
#define TPB     512

__device__ __forceinline__ unsigned pack2(float lo, float hi) {
    union { __bf16 h[2]; unsigned u; } c;
    c.h[0] = (__bf16)lo;
    c.h[1] = (__bf16)hi;
    return c.u;
}

// Build a B fragment (32x16, K-major per lane) from 16 contiguous f32 (fallback).
__device__ __forceinline__ v16bf load_conv16(const float* __restrict__ p) {
    const float4* q = (const float4*)p;
    float4 f0 = q[0], f1 = q[1], f2 = q[2], f3 = q[3];
    union { v16bf v; unsigned u[8]; } r;
    r.u[0] = pack2(f0.x, f0.y); r.u[1] = pack2(f0.z, f0.w);
    r.u[2] = pack2(f1.x, f1.y); r.u[3] = pack2(f1.z, f1.w);
    r.u[4] = pack2(f2.x, f2.y); r.u[5] = pack2(f2.z, f2.w);
    r.u[6] = pack2(f3.x, f3.y); r.u[7] = pack2(f3.z, f3.w);
    return r.v;
}

// B fragment straight from pre-converted bf16 W: two 16B loads, zero VALU.
__device__ __forceinline__ v16bf load_bf16x16(const unsigned short* __restrict__ p) {
    union { v16bf v; uint4 q[2]; } r;
    r.q[0] = *(const uint4*)p;
    r.q[1] = *(const uint4*)(p + 8);
    return r.v;
}

// A fragment (16x32 bf16): lane half lhi picks K {lhi*16..+15}
__device__ __forceinline__ v16bf load_afrag(const unsigned short* lds, int row, int koff) {
    const unsigned short* p = lds + row * APITCH + koff;
    union { v16bf v; uint4 q[2]; } r;
    r.q[0] = *(const uint4*)p;          // K = koff .. koff+7
    r.q[1] = *(const uint4*)(p + 8);    // K = koff+8 .. koff+15
    return r.v;
}

__device__ __forceinline__ v8f wmma_bf16(v16bf a, v16bf b, v8f c) {
    return __builtin_amdgcn_wmma_f32_16x16x32_bf16(false, a, false, b, (short)0, c,
                                                   false, false);
}

// Deterministic hash -> N(0,1) stand-in for jax.random.normal (unchecked path).
__device__ __forceinline__ float hash_normal(unsigned a, unsigned b) {
    unsigned x = a * 0x9E3779B9u + b * 0x85EBCA77u + 0x165667B1u;
    x ^= x >> 15; x *= 0x2C1B3C6Du; x ^= x >> 12; x *= 0x297A2D39u; x ^= x >> 15;
    unsigned y = x * 0x01000193u + 0x811C9DC5u;
    y ^= y >> 16; y *= 0x85EBCA6Bu; y ^= y >> 13;
    float u1 = ((x >> 8) + 1u) * (1.0f / 16777216.0f);
    float u2 = (y >> 8) * (1.0f / 16777216.0f);
    return sqrtf(-2.0f * __logf(u1)) * __cosf(6.28318530718f * u2);
}

// Pass 0: W (904x1024 f32) -> bf16, one uint4 (8 elems) per thread.
// 452 blocks * 256 threads * 8 = 925,696 = OUTC*HDIM exactly.
__global__ __launch_bounds__(256, 4)
void convert_W(const float* __restrict__ W, unsigned short* __restrict__ Wbf) {
    size_t i = ((size_t)blockIdx.x * 256 + threadIdx.x) * 8;
    float4 f0 = *(const float4*)(W + i);
    float4 f1 = *(const float4*)(W + i + 4);
    union { unsigned u[4]; uint4 q; } r;
    r.u[0] = pack2(f0.x, f0.y); r.u[1] = pack2(f0.z, f0.w);
    r.u[2] = pack2(f1.x, f1.y); r.u[3] = pack2(f1.z, f1.w);
    *(uint4*)(Wbf + i) = r.q;
}

template <bool PRE>
__global__ __launch_bounds__(TPB, 1)
void svg_mdn_fused(const float* __restrict__ A,            // (S*B, 1024)
                   const float* __restrict__ W,            // (904, 1024) f32
                   const unsigned short* __restrict__ Wbf, // (904, 1024) bf16 (PRE)
                   const float* __restrict__ bias,         // (904,)
                   float* __restrict__ out) {              // (S*B, 10)
    __shared__ unsigned short abuf[2][ABUFSZ];             // 17 KB bf16 A stage x2
    __shared__ float          cbuf[BLOCK_M * CPITCH];      // 114 KB f32 logits

    const int tid  = threadIdx.x;
    const int wave = __builtin_amdgcn_readfirstlane(tid >> 5);
    const int lane = tid & 31;
    const int lhi  = lane >> 4;
    const int llo  = lane & 15;
    const long mbase = (long)blockIdx.x * BLOCK_M;

    // N-tile assignment: wave w owns tiles w, w+16, w+32, w+48, CLAMPED to 56.
    // Clamped waves recompute tile 56 and write bit-identical values to the
    // same LDS slots (benign duplicate) -> zero branches in the hot loop.
    int tb[4];
    unsigned wlane[4];  // loop-invariant per-lane W element offset (K excluded)
#pragma unroll
    for (int i = 0; i < 4; ++i) {
        int t = wave + NWAVES * i;
        if (t > NTILES - 1) t = NTILES - 1;
        tb[i] = t * 16;
        wlane[i] = (unsigned)(tb[i] + llo) * HDIM + (unsigned)(lhi * 16);
    }

    v8f acc[2][4] = {};

    // A staging mapping: 512 threads x 8 f32 = 32 rows x 128 K per chunk
    const int arow = tid >> 4;
    const int akp  = (tid & 15) * 8;
    const float* agp = A + (mbase + arow) * HDIM + akp;
    const int    aso = arow * APITCH + akp;                // LDS elem offset

    // Prologue: fetch chunk 0 into registers.
    float4 f0 = *(const float4*)(agp);
    float4 f1 = *(const float4*)(agp + 4);

    for (int c = 0; c < NCHUNK; ++c) {
        const int k0 = c * CHUNK_K;
        unsigned short* ab = abuf[c & 1];

        // Convert + store chunk c (registers were fetched last iteration).
        uint4 pk;
        pk.x = pack2(f0.x, f0.y); pk.y = pack2(f0.z, f0.w);
        pk.z = pack2(f1.x, f1.y); pk.w = pack2(f1.z, f1.w);
        *(uint4*)(ab + aso) = pk;

        // Issue chunk c+1 global loads now; latency hides behind the WMMAs.
        if (c + 1 < NCHUNK) {
            f0 = *(const float4*)(agp + k0 + CHUNK_K);
            f1 = *(const float4*)(agp + k0 + CHUNK_K + 4);
        }

        __syncthreads();   // buf[c&1] visible; also fences buf reuse (2-apart)

        // Uniform chunk base (SGPR) + invariant lane offset (VGPR) + imm kk.
        const unsigned short* __restrict__ wchunkb = PRE ? (Wbf + k0) : nullptr;
        const float*          __restrict__ wchunkf = PRE ? nullptr : (W + k0);

#pragma unroll
        for (int kk = 0; kk < CHUNK_K; kk += 32) {
            v16bf a0 = load_afrag(ab, llo,      kk + lhi * 16);
            v16bf a1 = load_afrag(ab, 16 + llo, kk + lhi * 16);
#pragma unroll
            for (int i = 0; i < 4; ++i) {
                v16bf bf = PRE ? load_bf16x16(wchunkb + wlane[i] + kk)
                               : load_conv16 (wchunkf + wlane[i] + kk);
                acc[0][i] = wmma_bf16(a0, bf, acc[0][i]);
                acc[1][i] = wmma_bf16(a1, bf, acc[1][i]);
            }
        }
    }

    // Epilogue: + bias, scatter C tiles into LDS
#pragma unroll
    for (int i = 0; i < 4; ++i) {
        int col  = tb[i] + llo;
        float bv = (col < OUTC) ? bias[col] : 0.0f;
#pragma unroll
        for (int mt = 0; mt < 2; ++mt) {
#pragma unroll
            for (int v = 0; v < 8; ++v) {
                int rl = mt * 16 + lhi * 8 + v;
                cbuf[rl * CPITCH + col] = acc[mt][i][v] + bv;
            }
        }
    }
    __syncthreads();

    // Sampling epilogue: 16 threads per row, 32 rows
    const int r = tid >> 4;
    const int j = tid & 15;
    const float* crow = cbuf + r * CPITCH;
    const long rg = mbase + r;
    if (j < 6) {
        // temp 1e-4 categorical over normalized logmix ~= argmax of raw logmix
        const int base = 4 + 150 * j;
        int best = 0; float bv = crow[base];
        for (int k = 1; k < NMIX; ++k) {
            float v = crow[base + k];
            if (v > bv) { bv = v; best = k; }
        }
        float mean = crow[base + NMIX + best];
        float ls   = crow[base + 2 * NMIX + best];
        float z    = hash_normal((unsigned)rg, (unsigned)j);
        out[rg * 10 + 4 + j] = mean + __expf(ls) * 0.01f * z;   // sqrt(1e-4)
    } else if (j == 6) {
        int best = 0; float bv = crow[0];
        for (int c = 1; c < 4; ++c) {
            float v = crow[c];
            if (v > bv) { bv = v; best = c; }
        }
        out[rg * 10 + 0] = (best == 0) ? 1.0f : 0.0f;
        out[rg * 10 + 1] = (best == 1) ? 1.0f : 0.0f;
        out[rg * 10 + 2] = (best == 2) ? 1.0f : 0.0f;
        out[rg * 10 + 3] = (best == 3) ? 1.0f : 0.0f;
    }
}

extern "C" void kernel_launch(void* const* d_in, const int* in_sizes, int n_in,
                              void* d_out, int out_size, void* d_ws, size_t ws_size,
                              hipStream_t stream) {
    const float* A    = (const float*)d_in[0];   // decoder_output (51,2048,1024)
    const float* W    = (const float*)d_in[1];   // (904,1024)
    const float* bias = (const float*)d_in[2];   // (904,)
    float* out = (float*)d_out;                  // (51,2048,10)

    const int  Mtot   = SDIM * BDIM;             // 104448, divisible by 32
    const dim3 grid(Mtot / BLOCK_M);             // 3264 workgroups
    const size_t wbytes = (size_t)OUTC * HDIM * sizeof(unsigned short); // 1.85MB

    if (ws_size >= wbytes) {
        unsigned short* Wbf = (unsigned short*)d_ws;
        convert_W<<<dim3(452), 256, 0, stream>>>(W, Wbf);   // 452*256*8 = OUTC*HDIM
        svg_mdn_fused<true><<<grid, TPB, 0, stream>>>(A, W, Wbf, bias, out);
    } else {
        svg_mdn_fused<false><<<grid, TPB, 0, stream>>>(A, W, nullptr, bias, out);
    }
}
